// TransformerEncoder_36275293782328
// MI455X (gfx1250) — matrix operand order repaired and checked by reading, not compiled, not run on hardware
//
#include <hip/hip_runtime.h>
#include <math.h>

#define DEV __device__ __forceinline__

// ---------- problem constants ----------
#define BB 16
#define NN 1024
#define DD 384
#define HH 6
#define DH 64
#define MM (BB * NN)   // 16384 token rows

typedef __attribute__((ext_vector_type(16))) __bf16 v16bf;
typedef __attribute__((ext_vector_type(8)))  float  v8f;

union Frag {
  v16bf v;
  uint4 u[2];
};

DEV unsigned short f2bf(float f) {
  unsigned u = __float_as_uint(f);
  u += 0x7FFFu + ((u >> 16) & 1u);   // round-to-nearest-even
  return (unsigned short)(u >> 16);
}
DEV unsigned pack2bf(float a, float b) {
  return (unsigned)f2bf(a) | ((unsigned)f2bf(b) << 16);
}
DEV v8f vzero() {
  v8f z;
#pragma unroll
  for (int i = 0; i < 8; ++i) z[i] = 0.0f;
  return z;
}

// Async global->LDS copy (ASYNCcnt path). VDST = VGPR holding the LDS byte
// address (low 32 bits of the flat shared address ARE the LDS offset per
// CDNA5 ISA 10.2); VADDR = 64-bit global address, GV mode.
DEV void async_ld_b128(const void* smem_dst, const void* gsrc) {
  unsigned lds = (unsigned)(uintptr_t)smem_dst;
  asm volatile("global_load_async_to_lds_b128 %0, %1, off"
               :: "v"(lds), "v"(gsrc) : "memory");
}
DEV void wait_async0() {
  asm volatile("s_wait_asynccnt 0x0" ::: "memory");
}

// ---------- f32 -> bf16 weight conversion ----------
__global__ void cvt_bf16_kernel(const float* __restrict__ src,
                                unsigned short* __restrict__ dst, int n) {
  int i = blockIdx.x * 256 + threadIdx.x;
  if (i < n) dst[i] = f2bf(src[i]);
}

// ---------- LayerNorm over D=384; bf16 out (+ optional f32 out) ----------
__global__ __launch_bounds__(128) void layernorm_kernel(
    const float* __restrict__ in, const float* __restrict__ w,
    const float* __restrict__ b, unsigned short* __restrict__ outb,
    float* __restrict__ outf) {
  const int row = blockIdx.x;
  const int t = threadIdx.x;
  const float* x = in + (size_t)row * DD;
  float v0 = x[t], v1 = x[t + 128], v2 = x[t + 256];
  float s = v0 + v1 + v2;
  float s2 = v0 * v0 + v1 * v1 + v2 * v2;
#pragma unroll
  for (int off = 16; off >= 1; off >>= 1) {
    s  += __shfl_xor(s,  off);
    s2 += __shfl_xor(s2, off);
  }
  __shared__ float ls[4], ls2[4];
  const int wave = t >> 5, lane = t & 31;
  if (lane == 0) { ls[wave] = s; ls2[wave] = s2; }
  __syncthreads();
  s  = ls[0]  + ls[1]  + ls[2]  + ls[3];
  s2 = ls2[0] + ls2[1] + ls2[2] + ls2[3];
  const float mean = s * (1.0f / DD);
  const float var  = s2 * (1.0f / DD) - mean * mean;
  const float rstd = rsqrtf(var + 1e-5f);
  unsigned short* ob = outb + (size_t)row * DD;
  float* of = outf ? outf + (size_t)row * DD : nullptr;
#pragma unroll
  for (int j = 0; j < 3; ++j) {
    int idx = t + 128 * j;
    float vv = (j == 0 ? v0 : (j == 1 ? v1 : v2));
    float y = (vv - mean) * rstd * w[idx] + b[idx];
    ob[idx] = f2bf(y);
    if (of) of[idx] = y;
  }
}

// ---------- generic bf16 WMMA GEMM: C[M,Nout] = A[M,K] * Bw[Nout,K]^T ----------
// 256 thr = 8 waves (4 M x 2 N); block tile 128x128; wave tile 32x64.
// MODE 0: bf16 out = acc*scale             (Q,K projections)
// MODE 1: bf16 V-transposed out [b,h,kk,n] (V projection)
// MODE 2: f32 out = acc + bias + resid     (Wo proj / MLP2, fused residual)
// MODE 3: bf16 out = gelu(acc + bias)      (MLP1)
template <int MODE>
__global__ __launch_bounds__(256) void gemm_bf16(
    const unsigned short* __restrict__ A, const unsigned short* __restrict__ Bw,
    int K, int Nout, unsigned short* __restrict__ outb,
    float* __restrict__ outf, const float* __restrict__ bias,
    const float* __restrict__ resid, float scale) {
  const int lane = threadIdx.x & 31;
  const int wave = threadIdx.x >> 5;
  const int wm = wave & 3, wn = wave >> 2;
  const int m0 = blockIdx.y * 128 + wm * 32;
  const int n0 = blockIdx.x * 128 + wn * 64;
  const int r = lane & 15, kh = lane >> 4;

  v8f acc[2][4];
#pragma unroll
  for (int mt = 0; mt < 2; ++mt)
#pragma unroll
    for (int nt = 0; nt < 4; ++nt) acc[mt][nt] = vzero();

  for (int k0 = 0; k0 < K; k0 += 32) {
    Frag a[2], bf[4];
#pragma unroll
    for (int mt = 0; mt < 2; ++mt) {
      const unsigned short* p = A + (size_t)(m0 + 16 * mt + r) * K + k0;
      a[mt].u[0] = *(const uint4*)(p + 16 * kh);
      a[mt].u[1] = *(const uint4*)(p + 16 * kh + 8);
    }
#pragma unroll
    for (int nt = 0; nt < 4; ++nt) {
      const unsigned short* p = Bw + (size_t)(n0 + 16 * nt + r) * K + k0 + 16 * kh;
      bf[nt].u[0] = *(const uint4*)(p);
      bf[nt].u[1] = *(const uint4*)(p + 8);
    }
#pragma unroll
    for (int mt = 0; mt < 2; ++mt)
#pragma unroll
      for (int nt = 0; nt < 4; ++nt)
        acc[mt][nt] = __builtin_amdgcn_wmma_f32_16x16x32_bf16(
            false, a[mt].v, false, bf[nt].v, (short)0, acc[mt][nt], false, false);
  }

#pragma unroll
  for (int mt = 0; mt < 2; ++mt)
#pragma unroll
    for (int nt = 0; nt < 4; ++nt)
#pragma unroll
      for (int i = 0; i < 8; ++i) {
        const int grow = m0 + 16 * mt + i + 8 * kh;
        const int gcol = n0 + 16 * nt + r;
        const float v = acc[mt][nt][i];
        if (MODE == 0) {
          outb[(size_t)grow * Nout + gcol] = f2bf(v * scale);
        } else if (MODE == 1) {
          const int bb2 = grow >> 10, n = grow & 1023;
          const int h = gcol >> 6, kk = gcol & 63;
          outb[((size_t)((bb2 * HH + h) * DH + kk) << 10) + n] = f2bf(v);
        } else if (MODE == 2) {
          const size_t idx = (size_t)grow * Nout + gcol;
          outf[idx] = v + bias[gcol] + resid[idx];
        } else {
          const float x = v + bias[gcol];
          const float g = 0.5f * x * (1.0f + erff(x * 0.70710678118654752f));
          outb[(size_t)grow * Nout + gcol] = f2bf(g);
        }
      }
}

// ---------- flash attention (S^T = K*Q^T trick; softmax stats lane-local) ----
// grid: (N/128, B*H); 8 waves/block; wave owns 16 queries.
// K/V^T tiles staged cooperatively in LDS via async global->LDS DMA, double
// buffered; fragments read from LDS with padded strides (bank-conflict free).
#define KPAD 72   // 64 + 8 bf16 elements (row = 36 dwords)
#define VPAD 40   // 32 + 8 bf16 elements (row = 20 dwords)

__global__ __launch_bounds__(256) void flash_attn_kernel(
    const unsigned short* __restrict__ Qb, const unsigned short* __restrict__ Kb,
    const unsigned short* __restrict__ VT, unsigned short* __restrict__ O) {
  __shared__ __align__(16) unsigned short sK[2][32 * KPAD];  // 2 x 4.5KB
  __shared__ __align__(16) unsigned short sV[2][64 * VPAD];  // 2 x 5KB

  const int tid = threadIdx.x;
  const int lane = tid & 31;
  const int wave = tid >> 5;
  const int bh = blockIdx.y;
  const int b = bh / HH, h = bh % HH;
  const int n0 = blockIdx.x * 128 + wave * 16;
  const int r = lane & 15, kh = lane >> 4;
  const bool lo16 = (lane < 16);

  // Q^T B-fragments (held for whole pass): column n = lane, dh contiguous
  Frag qf[2];
  {
    const unsigned short* qrow =
        Qb + (size_t)(b * NN + n0 + r) * DD + h * DH + 16 * kh;
#pragma unroll
    for (int j = 0; j < 2; ++j) {
      qf[j].u[0] = *(const uint4*)(qrow + 32 * j);
      qf[j].u[1] = *(const uint4*)(qrow + 32 * j + 8);
    }
  }

  const unsigned short* kbase = Kb + (size_t)(b * NN) * DD + h * DH;
  const unsigned short* vbase = VT + ((size_t)(b * HH + h) << 16);  // *64*1024

  // per-thread async-copy slot: K chunk 32x64 (8 x 16B per row),
  // V^T chunk 64x32 (4 x 16B per row); 16B per thread per tile.
  const int ckr = tid >> 3, ckc = (tid & 7) * 8;
  const int cvr = tid >> 2, cvc = (tid & 3) * 8;

  auto issue = [&](int m0, int buf) {
    async_ld_b128(&sK[buf][ckr * KPAD + ckc],
                  kbase + (size_t)(m0 + ckr) * DD + ckc);
    async_ld_b128(&sV[buf][cvr * VPAD + cvc],
                  vbase + ((size_t)cvr << 10) + m0 + cvc);
  };

  issue(0, 0);

  v8f o_acc[4];
#pragma unroll
  for (int kt = 0; kt < 4; ++kt) o_acc[kt] = vzero();
  float mx = -1e30f, l = 0.0f;
  const v8f zero = vzero();

  for (int it = 0; it < NN / 32; ++it) {
    const int cur = it & 1;
    wait_async0();        // this wave's LDS writes for `cur` have landed
    __syncthreads();      // ...and everyone else's too
    if (it + 1 < NN / 32) issue((it + 1) * 32, cur ^ 1);  // overlap with math

    // S^T(32 keys x 16 queries) = K * Q^T, fragments from LDS
    v8f S[2];
#pragma unroll
    for (int s = 0; s < 2; ++s) {
      const unsigned short* krow = &sK[cur][(16 * s + r) * KPAD];
      Frag kf0, kf1;
      kf0.u[0] = *(const uint4*)(krow + 16 * kh);
      kf0.u[1] = *(const uint4*)(krow + 16 * kh + 8);
      kf1.u[0] = *(const uint4*)(krow + 32 + 16 * kh);
      kf1.u[1] = *(const uint4*)(krow + 40 + 16 * kh);
      v8f t0 = __builtin_amdgcn_wmma_f32_16x16x32_bf16(
          false, kf0.v, false, qf[0].v, (short)0, zero, false, false);
      S[s] = __builtin_amdgcn_wmma_f32_16x16x32_bf16(
          false, kf1.v, false, qf[1].v, (short)0, t0, false, false);
    }

    // online softmax: per-query stats are lane-local + one half-wave swap
    float tmax = S[0][0];
#pragma unroll
    for (int i = 0; i < 8; ++i) {
      tmax = fmaxf(tmax, S[0][i]);
      tmax = fmaxf(tmax, S[1][i]);
    }
    tmax = fmaxf(tmax, __shfl_xor(tmax, 16));
    const float nm = fmaxf(mx, tmax);
    const float alpha = __expf(mx - nm);
    mx = nm;
    float rs = 0.0f;
#pragma unroll
    for (int s = 0; s < 2; ++s)
#pragma unroll
      for (int i = 0; i < 8; ++i) {
        const float e = __expf(S[s][i] - nm);
        S[s][i] = e;
        rs += e;
      }
    rs += __shfl_xor(rs, 16);
    l = l * alpha + rs;
#pragma unroll
    for (int kt = 0; kt < 4; ++kt)
#pragma unroll
      for (int i = 0; i < 8; ++i) o_acc[kt][i] *= alpha;

    // repack P^T (C-layout f32) into a B-fragment (bf16): one shfl_xor(16)/dword
    unsigned pd[8];
#pragma unroll
    for (int d = 0; d < 8; ++d) {
      const int rr = (d < 4) ? 2 * d : 2 * d - 8;
      const float x0o = (d < 4) ? S[0][rr]     : S[1][rr];
      const float x1o = (d < 4) ? S[0][rr + 1] : S[1][rr + 1];
      const float x0x = __shfl_xor((d < 4) ? S[1][rr]     : S[0][rr],     16);
      const float x1x = __shfl_xor((d < 4) ? S[1][rr + 1] : S[0][rr + 1], 16);
      const bool own = (d < 4) ? lo16 : !lo16;
      pd[d] = pack2bf(own ? x0o : x0x, own ? x1o : x1x);
    }
    Frag pf;
    pf.u[0] = make_uint4(pd[0], pd[1], pd[2], pd[3]);
    pf.u[1] = make_uint4(pd[4], pd[5], pd[6], pd[7]);

    // O^T += V^T * P^T (A = V^T fragments from LDS)
#pragma unroll
    for (int kt = 0; kt < 4; ++kt) {
      const unsigned short* vrow = &sV[cur][(16 * kt + r) * VPAD];
      Frag vf;
      vf.u[0] = *(const uint4*)(vrow + 16 * kh);
      vf.u[1] = *(const uint4*)(vrow + 16 * kh + 8);
      o_acc[kt] = __builtin_amdgcn_wmma_f32_16x16x32_bf16(
          false, vf.v, false, pf.v, (short)0, o_acc[kt], false, false);
    }
  }

  const float invl = 1.0f / l;
  unsigned short* orow = O + (size_t)(b * NN + n0 + r) * DD + h * DH;
#pragma unroll
  for (int kt = 0; kt < 4; ++kt)
#pragma unroll
    for (int i = 0; i < 8; ++i)
      orow[16 * kt + i + 8 * kh] = f2bf(o_acc[kt][i] * invl);
}

// ---------- host-side launcher ----------
extern "C" void kernel_launch(void* const* d_in, const int* in_sizes, int n_in,
                              void* d_out, int out_size, void* d_ws,
                              size_t ws_size, hipStream_t stream) {
  (void)in_sizes; (void)n_in; (void)out_size; (void)ws_size;
  const float* X     = (const float*)d_in[0];
  const float* Wq    = (const float*)d_in[1];
  const float* Wk    = (const float*)d_in[2];
  const float* Wv    = (const float*)d_in[3];
  const float* Wo    = (const float*)d_in[4];
  const float* bo    = (const float*)d_in[5];
  const float* ln1w  = (const float*)d_in[6];
  const float* ln1b  = (const float*)d_in[7];
  const float* ln2w  = (const float*)d_in[8];
  const float* ln2b  = (const float*)d_in[9];
  const float* W1    = (const float*)d_in[10];
  const float* b1    = (const float*)d_in[11];
  const float* W2    = (const float*)d_in[12];
  const float* b2    = (const float*)d_in[13];
  float* out         = (float*)d_out;

  size_t off = 0;
  auto alloc = [&](size_t bytes) {
    void* p = (char*)d_ws + off;
    off += (bytes + 255) & ~(size_t)255;
    return p;
  };
  unsigned short* t_bf = (unsigned short*)alloc((size_t)MM * DD * 2);
  unsigned short* Qb   = (unsigned short*)alloc((size_t)MM * DD * 2);
  unsigned short* Kb   = (unsigned short*)alloc((size_t)MM * DD * 2);
  unsigned short* VT   = (unsigned short*)alloc((size_t)MM * DD * 2);
  unsigned short* Ob   = (unsigned short*)alloc((size_t)MM * DD * 2);
  float*          Y    = (float*)alloc((size_t)MM * DD * 4);
  float*          X2f  = (float*)alloc((size_t)MM * DD * 4);
  unsigned short* X2b  = (unsigned short*)alloc((size_t)MM * DD * 2);
  unsigned short* H1   = (unsigned short*)alloc((size_t)MM * 2 * DD * 2);
  unsigned short* wqb  = (unsigned short*)alloc((size_t)DD * DD * 2);
  unsigned short* wkb  = (unsigned short*)alloc((size_t)DD * DD * 2);
  unsigned short* wvb  = (unsigned short*)alloc((size_t)DD * DD * 2);
  unsigned short* wob  = (unsigned short*)alloc((size_t)DD * DD * 2);
  unsigned short* w1b  = (unsigned short*)alloc((size_t)2 * DD * DD * 2);
  unsigned short* w2b  = (unsigned short*)alloc((size_t)2 * DD * DD * 2);

  const int nw = DD * DD;        // 147456
  const int nw2 = 2 * DD * DD;   // 294912
  cvt_bf16_kernel<<<(nw  + 255) / 256, 256, 0, stream>>>(Wq, wqb, nw);
  cvt_bf16_kernel<<<(nw  + 255) / 256, 256, 0, stream>>>(Wk, wkb, nw);
  cvt_bf16_kernel<<<(nw  + 255) / 256, 256, 0, stream>>>(Wv, wvb, nw);
  cvt_bf16_kernel<<<(nw  + 255) / 256, 256, 0, stream>>>(Wo, wob, nw);
  cvt_bf16_kernel<<<(nw2 + 255) / 256, 256, 0, stream>>>(W1, w1b, nw2);
  cvt_bf16_kernel<<<(nw2 + 255) / 256, 256, 0, stream>>>(W2, w2b, nw2);

  // LN1 -> t (bf16)
  layernorm_kernel<<<MM, 128, 0, stream>>>(X, ln1w, ln1b, t_bf, nullptr);

  const dim3 g384(3, MM / 128);  // Nout=384
  const dim3 g768(6, MM / 128);  // Nout=768

  // QKV projections (Q pre-scaled by 1/sqrt(64)); V stored transposed
  gemm_bf16<0><<<g384, 256, 0, stream>>>(t_bf, wqb, DD, DD, Qb, nullptr,
                                         nullptr, nullptr, 0.125f);
  gemm_bf16<0><<<g384, 256, 0, stream>>>(t_bf, wkb, DD, DD, Kb, nullptr,
                                         nullptr, nullptr, 1.0f);
  gemm_bf16<1><<<g384, 256, 0, stream>>>(t_bf, wvb, DD, DD, VT, nullptr,
                                         nullptr, nullptr, 1.0f);

  // flash attention
  flash_attn_kernel<<<dim3(NN / 128, BB * HH), 256, 0, stream>>>(Qb, Kb, VT, Ob);

  // output projection + bias + residual (f32)
  gemm_bf16<2><<<g384, 256, 0, stream>>>(Ob, wob, DD, DD, nullptr, Y, bo, X,
                                         1.0f);

  // LN2 -> X2 (f32 for final residual, bf16 for MLP GEMM)
  layernorm_kernel<<<MM, 128, 0, stream>>>(Y, ln2w, ln2b, X2b, X2f);

  // MLP: W1 + exact GELU -> bf16, then W2 + bias + residual -> d_out (f32)
  gemm_bf16<3><<<g768, 256, 0, stream>>>(X2b, w1b, DD, 2 * DD, H1, nullptr, b1,
                                         nullptr, 1.0f);
  gemm_bf16<2><<<g384, 256, 0, stream>>>(H1, w2b, 2 * DD, DD, nullptr, out, b2,
                                         X2f, 1.0f);
}
